// MutiAtten_3899830305375
// MI455X (gfx1250) — matrix operand order repaired and checked by reading, not compiled, not run on hardware
//
#include <hip/hip_runtime.h>
#include <stdint.h>

#ifndef __has_builtin
#define __has_builtin(x) 0
#endif

// gfx1250 async global->LDS path (ASYNCcnt-tracked), with safe fallback.
#if __has_builtin(__builtin_amdgcn_global_load_async_to_lds_b128)
#define HAVE_ASYNC_LDS 1
#else
#define HAVE_ASYNC_LDS 0
#endif

#define AS_GLOBAL __attribute__((address_space(1)))
#define AS_LDS    __attribute__((address_space(3)))

typedef __attribute__((ext_vector_type(16))) __bf16 v16bf;
typedef __attribute__((ext_vector_type(8)))  float  v8f;
typedef __attribute__((ext_vector_type(4)))  int    v4i;

static constexpr int LSEQ = 2048;   // sequence length
static constexpr int NDIM = 1024;   // feature dim (no head split in reference)
static constexpr int BM = 128, BN = 128, BK = 32;
static constexpr int KP = 40;       // NT tile K-stride (80B: 16B-aligned, 20-bank stagger)
static constexpr int NP = 136;      // NN (V) tile N-stride (272B: 16B-aligned)

__device__ __forceinline__ uint16_t f32_bf16(float f) {
  uint32_t u = __float_as_uint(f);
  return (uint16_t)((u + 0x7FFFu + ((u >> 16) & 1u)) >> 16); // RNE
}
__device__ __forceinline__ uint32_t pack2_bf16(float x, float y) {
  return (uint32_t)f32_bf16(x) | ((uint32_t)f32_bf16(y) << 16);
}

// 16B global -> LDS, async on CDNA5 (global_load_async_to_lds_b128).
// Builtin prototype (probe-confirmed): (int4 AS1*, int4 AS3*, imm offset, imm cpol).
__device__ __forceinline__ void copy16_async(const uint16_t* g, uint16_t* l) {
#if HAVE_ASYNC_LDS
  __builtin_amdgcn_global_load_async_to_lds_b128(
      (AS_GLOBAL v4i*)g, (AS_LDS v4i*)l, 0, 0);
#else
  *(uint4*)l = *(const uint4*)g;
#endif
}
__device__ __forceinline__ void wait_async_lds() {
#if HAVE_ASYNC_LDS
#if __has_builtin(__builtin_amdgcn_s_wait_asynccnt)
  __builtin_amdgcn_s_wait_asynccnt(0);
#else
  asm volatile("s_wait_asynccnt 0x0" ::: "memory");
#endif
#endif
}

// 8 consecutive f32 -> 8 bf16, one 16B LDS store.
__device__ __forceinline__ void cvt8_store(const float* g, uint16_t* l) {
  const float4* gp = (const float4*)g;
  const float4 a = gp[0], b = gp[1];
  uint4 p;
  p.x = pack2_bf16(a.x, a.y);
  p.y = pack2_bf16(a.z, a.w);
  p.z = pack2_bf16(b.x, b.y);
  p.w = pack2_bf16(b.z, b.w);
  *(uint4*)l = p;
}

union AccU  { v8f  v; float    f[8]; };
union FragU { v16bf v; uint32_t u[8]; };

// MODE 0: kqv GEMM   A=X(f32) B=W_kqv(f32,NT) +bias -> split K/Q/V bf16 (Q pre-scaled)
// MODE 1: scores     A=Q(bf16) B=K(bf16,NT)          -> S f32 (per batch, blockIdx.z)
// MODE 2: attn out   A=P(bf16) B=V(bf16,NN)          -> O^T bf16 (feature-major store)
// MODE 3: projection A=O^T-view(bf16) B=W_proj(f32,NT) +bias -> out f32
template<int MODE>
__global__ __launch_bounds__(256)
void gemm_wmma_kernel(const float* __restrict__ Af, const uint16_t* __restrict__ Ab,
                      const float* __restrict__ Bf, const uint16_t* __restrict__ Bb,
                      const float* __restrict__ bias,
                      float* __restrict__ Of, uint16_t* __restrict__ Obf,
                      uint16_t* __restrict__ Ok, uint16_t* __restrict__ Oq,
                      uint16_t* __restrict__ Ov,
                      int M, int N, int K, float qscale)
{
  constexpr int ATILE = BM * KP;
  constexpr int BTILE = (MODE == 2) ? (BK * NP) : (BN * KP);
  alignas(16) __shared__ uint16_t As[2 * ATILE];   // double-buffered
  alignas(16) __shared__ uint16_t Bs[2 * BTILE];

  const int tid  = threadIdx.x;
  const int lane = tid & 31;
  const int wave = tid >> 5;          // 8 waves, 2x4 grid, 64x32 per wave
  const int wm   = (wave >> 2) * 64;
  const int wn   = (wave & 3) * 32;
  const int half = lane >> 4;
  const int l16  = lane & 15;

  size_t zA = 0, zB = 0, zO = 0;
  if constexpr (MODE == 1) {
    const size_t z = blockIdx.z;
    zA = z * (size_t)LSEQ * NDIM; zB = z * (size_t)LSEQ * NDIM; zO = z * (size_t)LSEQ * LSEQ;
  }
  if constexpr (MODE == 2) {
    const size_t z = blockIdx.z;
    zA = z * (size_t)LSEQ * LSEQ; zB = z * (size_t)LSEQ * NDIM; zO = z * (size_t)LSEQ * NDIM;
  }

  // Stage loader: 512 units of 8 bf16 (16B) per tile, 2 units/thread.
  auto load_stage = [&](int k0, int buf) {
    #pragma unroll
    for (int it = 0; it < 2; ++it) {
      const int u = tid + it * 256;
      const int r = u >> 2, cu = (u & 3) * 8;
      const size_t g = zA + (size_t)(blockIdx.y * BM + r) * K + (size_t)(k0 + cu);
      uint16_t* l = &As[buf * ATILE + r * KP + cu];
      if constexpr (MODE == 0) cvt8_store(Af + g, l);
      else                     copy16_async(Ab + g, l);
    }
    #pragma unroll
    for (int it = 0; it < 2; ++it) {
      const int u = tid + it * 256;
      if constexpr (MODE == 2) {
        // NN operand kept [k][n]; async rows of V straight into LDS
        const int kk = u >> 4, nu = (u & 15) * 8;
        const size_t g = zB + (size_t)(k0 + kk) * N + (size_t)(blockIdx.x * BN + nu);
        copy16_async(Bb + g, &Bs[buf * BTILE + kk * NP + nu]);
      } else {
        const int r = u >> 2, cu = (u & 3) * 8;
        const size_t g = zB + (size_t)(blockIdx.x * BN + r) * K + (size_t)(k0 + cu);
        uint16_t* l = &Bs[buf * BTILE + r * KP + cu];
        if constexpr (MODE == 0 || MODE == 3) cvt8_store(Bf + g, l);
        else                                  copy16_async(Bb + g, l);
      }
    }
  };

  AccU acc[4][2];
  #pragma unroll
  for (int a = 0; a < 4; ++a)
    #pragma unroll
    for (int b = 0; b < 2; ++b)
      #pragma unroll
      for (int i = 0; i < 8; ++i) acc[a][b].f[i] = 0.f;

  load_stage(0, 0);
  const int NSTAGE = K / BK;
  for (int s = 0; s < NSTAGE; ++s) {
    const int buf = s & 1;
    wait_async_lds();     // my stage-s async copies have landed in LDS
    __syncthreads();      // (includes dscnt wait) everyone's copies visible,
                          // and everyone finished reading buf^1 from stage s-1
    if (s + 1 < NSTAGE) load_stage((s + 1) * BK, buf ^ 1);

    // ---- fragment gather, contiguous k = 16*half + s for A(16x32) and B(32x16) ----
    FragU afr[4], bfr[2];
    const uint16_t* Abase = &As[buf * ATILE];
    #pragma unroll
    for (int fm = 0; fm < 4; ++fm) {
      const uint16_t* base = Abase + (wm + fm * 16 + l16) * KP;
      #pragma unroll
      for (int i = 0; i < 8; ++i) {
        const int kk = half * 16 + i * 2;
        afr[fm].u[i] = *(const uint32_t*)(base + kk);
      }
    }
    const uint16_t* Bbase = &Bs[buf * BTILE];
    if constexpr (MODE == 2) {
      #pragma unroll
      for (int fn = 0; fn < 2; ++fn) {
        const int n = wn + fn * 16 + l16;
        #pragma unroll
        for (int i = 0; i < 8; ++i) {
          const int k = half * 16 + i * 2;
          const uint32_t lo = Bbase[k * NP + n];
          const uint32_t hi = Bbase[(k + 1) * NP + n];
          bfr[fn].u[i] = lo | (hi << 16);
        }
      }
    } else {
      #pragma unroll
      for (int fn = 0; fn < 2; ++fn) {
        const uint16_t* base = Bbase + (wn + fn * 16 + l16) * KP + half * 16;
        #pragma unroll
        for (int i = 0; i < 8; ++i)
          bfr[fn].u[i] = *(const uint32_t*)(base + i * 2);
      }
    }
    #pragma unroll
    for (int fm = 0; fm < 4; ++fm)
      #pragma unroll
      for (int fn = 0; fn < 2; ++fn)
        acc[fm][fn].v = __builtin_amdgcn_wmma_f32_16x16x32_bf16(
            false, afr[fm].v, false, bfr[fn].v, (short)0, acc[fm][fn].v,
            false, false);
  }

  // ---- epilogue: D layout -> VGPR i holds row i+8*(lane>>4), col lane&15 ----
  const int rowBase = blockIdx.y * BM + wm;
  const int colBase = blockIdx.x * BN + wn;
  #pragma unroll
  for (int fm = 0; fm < 4; ++fm)
    #pragma unroll
    for (int fn = 0; fn < 2; ++fn)
      #pragma unroll
      for (int i = 0; i < 8; ++i) {
        const int row = rowBase + fm * 16 + i + 8 * half;
        const int col = colBase + fn * 16 + l16;
        float v = acc[fm][fn].f[i];
        if constexpr (MODE == 0) {
          v += bias[col];
          const int seg = col >> 10, cc = col & 1023; // torch split order: k,q,v
          const uint16_t hv = f32_bf16(seg == 1 ? v * qscale : v);
          uint16_t* dst = (seg == 0) ? Ok : (seg == 1) ? Oq : Ov;
          dst[(size_t)row * NDIM + cc] = hv;
        } else if constexpr (MODE == 1) {
          Of[zO + (size_t)row * N + col] = v;                 // scores f32
        } else if constexpr (MODE == 2) {
          Obf[zO + (size_t)col * LSEQ + row] = f32_bf16(v);   // O^T (replicates view bug)
        } else {
          Of[(size_t)row * N + col] = v + bias[col];
        }
      }
}

// One block per (query row, batch): masked softmax in f32, write P as bf16.
// exp(-1e6 - max) underflows to exactly 0.f, matching reference semantics.
__global__ __launch_bounds__(256)
void softmax_mask_kernel(const float* __restrict__ S, uint16_t* __restrict__ P,
                         const int* __restrict__ valid_lens)
{
  const int b = blockIdx.y, q = blockIdx.x;
  const int tid = threadIdx.x;
  const float* row = S + ((size_t)b * LSEQ + q) * LSEQ;
  uint16_t* prow   = P + ((size_t)b * LSEQ + q) * LSEQ;
  const int valid = valid_lens[b];

  __shared__ float red[256];
  float vals[LSEQ / 256];
  float mx = -3.0e38f;
  #pragma unroll
  for (int t = 0; t < LSEQ / 256; ++t) {
    const int j = tid + t * 256;
    float v = row[j];
    if (j >= valid) v = -1000000.0f;   // MASK_FILL
    vals[t] = v;
    mx = fmaxf(mx, v);
  }
  red[tid] = mx; __syncthreads();
  for (int s = 128; s > 0; s >>= 1) {
    if (tid < s) red[tid] = fmaxf(red[tid], red[tid + s]);
    __syncthreads();
  }
  mx = red[0]; __syncthreads();

  float sum = 0.f;
  #pragma unroll
  for (int t = 0; t < LSEQ / 256; ++t) {
    vals[t] = __expf(vals[t] - mx);
    sum += vals[t];
  }
  red[tid] = sum; __syncthreads();
  for (int s = 128; s > 0; s >>= 1) {
    if (tid < s) red[tid] += red[tid + s];
    __syncthreads();
  }
  const float inv = 1.0f / red[0];
  #pragma unroll
  for (int t = 0; t < LSEQ / 256; ++t)
    prow[tid + t * 256] = f32_bf16(vals[t] * inv);
}

extern "C" void kernel_launch(void* const* d_in, const int* in_sizes, int n_in,
                              void* d_out, int out_size, void* d_ws, size_t ws_size,
                              hipStream_t stream)
{
  const float* pre   = (const float*)d_in[0];  // (B,L,D) f32
  const int*   vlen  = (const int*)  d_in[1];  // (B,) i32
  const float* Wkqv  = (const float*)d_in[2];  // (3D,D) f32
  const float* bkqv  = (const float*)d_in[3];  // (3D,)  f32
  const float* Wproj = (const float*)d_in[4];  // (D,D)  f32
  const float* bproj = (const float*)d_in[5];  // (D,)   f32
  float* out = (float*)d_out;                  // (B,L,D) f32

  // Workspace: K/Q/V bf16 (48MB) | S f32 (64MB) | P bf16 (32MB) | O^T bf16 (16MB)
  char* ws = (char*)d_ws;
  const size_t MB = 1024ull * 1024ull;
  uint16_t* Kb = (uint16_t*)(ws +   0 * MB);
  uint16_t* Qb = (uint16_t*)(ws +  16 * MB);
  uint16_t* Vb = (uint16_t*)(ws +  32 * MB);
  float*    S  = (float*)   (ws +  48 * MB);
  uint16_t* P  = (uint16_t*)(ws + 112 * MB);
  uint16_t* Yt = (uint16_t*)(ws + 144 * MB);

  const float qscale = 0.03125f; // 1/sqrt(1024), folded into Q at store

  // 1) kqv = X @ Wkqv^T + b  -> split K/Q/V bf16        (M=8192, N=3072, K=1024)
  gemm_wmma_kernel<0><<<dim3(3 * NDIM / BN, (4 * LSEQ) / BM, 1), 256, 0, stream>>>(
      pre, nullptr, Wkqv, nullptr, bkqv, nullptr, nullptr, Kb, Qb, Vb,
      4 * LSEQ, 3 * NDIM, NDIM, qscale);
  // 2) S = (Q*scale) @ K^T per batch                    (M=N=2048, K=1024)
  gemm_wmma_kernel<1><<<dim3(LSEQ / BN, LSEQ / BM, 4), 256, 0, stream>>>(
      nullptr, Qb, nullptr, Kb, nullptr, S, nullptr, nullptr, nullptr, nullptr,
      LSEQ, LSEQ, NDIM, 1.f);
  // 3) masked softmax rows -> P bf16
  softmax_mask_kernel<<<dim3(LSEQ, 4), 256, 0, stream>>>(S, P, vlen);
  // 4) O = P @ V, stored transposed (feature-major)     (M=2048, N=1024, K=2048)
  gemm_wmma_kernel<2><<<dim3(NDIM / BN, LSEQ / BM, 4), 256, 0, stream>>>(
      nullptr, P, nullptr, Vb, nullptr, nullptr, Yt, nullptr, nullptr, nullptr,
      LSEQ, NDIM, LSEQ, 1.f);
  // 5) out = view(O^T) @ Wproj^T + b                    (M=8192, N=1024, K=1024)
  gemm_wmma_kernel<3><<<dim3(NDIM / BN, (4 * LSEQ) / BM, 1), 256, 0, stream>>>(
      nullptr, Yt, Wproj, nullptr, bproj, out, nullptr, nullptr, nullptr, nullptr,
      4 * LSEQ, NDIM, NDIM, 1.f);

  (void)in_sizes; (void)n_in; (void)out_size; (void)ws_size;
}
